// ScaledDotProductAttention_1692217115311
// MI455X (gfx1250) — compile-verified, matrix operand order checked by reading, not run on hardware
//
#include <hip/hip_runtime.h>
#include <hip/hip_fp16.h>
#include <hip/hip_bf16.h>

// CDNA5 (gfx1250) one-pass attention:
//   scores = Q K^T (WMMA f16->f32), softmax in LDS, attention written once,
//   context = P V (WMMA) + residual Q.
// Wave32; each wave owns 16 query rows of one batch; no inter-wave sync.

typedef __attribute__((ext_vector_type(16))) _Float16 v16h;
typedef __attribute__((ext_vector_type(8)))  _Float16 v8h;
typedef __attribute__((ext_vector_type(8)))  float    v8f;

#define B_SZ 32
#define S_SZ 2048
#define D_SZ 64
#define WAVES_PER_WG 4
#define ROWSTRIDE 2056                      // 2048 + 8 halves pad: keeps 16B align, staggers banks
#define PROB_HALFS (16 * ROWSTRIDE)
#define LDS_BYTES (WAVES_PER_WG * PROB_HALFS * 2)   // 263168 B < 320 KB/WGP

// Build a 16-half A/B fragment for v_wmma_f32_16x16x32_f16 from an fp32 row.
// Per-lane layout (ISA 7.12.2, 16-bit A 16x32): lanes 0-15 hold K={0..7,16..23},
// lanes 16-31 hold K={8..15,24..31} of their row -> pass base = (chunk*32 + kb0).
__device__ __forceinline__ v16h load_frag_f32(const float* __restrict__ row, int base) {
    float4 x0 = *(const float4*)(row + base);
    float4 x1 = *(const float4*)(row + base + 4);
    float4 x2 = *(const float4*)(row + base + 16);
    float4 x3 = *(const float4*)(row + base + 20);
    v16h f;
    f[0]=(_Float16)x0.x;  f[1]=(_Float16)x0.y;  f[2]=(_Float16)x0.z;  f[3]=(_Float16)x0.w;
    f[4]=(_Float16)x1.x;  f[5]=(_Float16)x1.y;  f[6]=(_Float16)x1.z;  f[7]=(_Float16)x1.w;
    f[8]=(_Float16)x2.x;  f[9]=(_Float16)x2.y;  f[10]=(_Float16)x2.z; f[11]=(_Float16)x2.w;
    f[12]=(_Float16)x3.x; f[13]=(_Float16)x3.y; f[14]=(_Float16)x3.z; f[15]=(_Float16)x3.w;
    return f;
}

__global__ void attn_fused_kernel(const float* __restrict__ q,
                                  const float* __restrict__ k,
                                  const float* __restrict__ v,
                                  float* __restrict__ out) {
    extern __shared__ _Float16 smem[];
    const int lane   = threadIdx.x & 31;
    const int wave   = threadIdx.x >> 5;
    const int qblock = blockIdx.x * WAVES_PER_WG + wave;   // 0..4095
    const int b      = qblock >> 7;                        // / (S/16)
    const int qb     = qblock & 127;

    _Float16* probs = smem + wave * PROB_HALFS;            // per-wave 16 x 2048 (padded)

    const bool lo  = lane < 16;
    const int  ln  = lane & 15;
    const int  kb0 = lo ? 0 : 8;                           // K-subset base for this half-wave

    const float* Qb = q + (size_t)(b * S_SZ) * D_SZ;
    const float* Kb = k + (size_t)(b * S_SZ) * D_SZ;
    const float* Vb = v + (size_t)(b * S_SZ) * D_SZ;
    float*       ctx  = out;
    float*       attn = out + (size_t)B_SZ * S_SZ * D_SZ;

    // ---- Q tile as two A-fragments (K-dim = D = 64 -> 2 chunks of 32) ----
    const float* qrow = Qb + (size_t)(qb * 16 + ln) * D_SZ;
    v16h aq0 = load_frag_f32(qrow, kb0);
    v16h aq1 = load_frag_f32(qrow, 32 + kb0);

    // ---- Phase 1: raw scores -> LDS (f16) ----
    for (int kt = 0; kt < S_SZ / 16; ++kt) {
        const float* krow = Kb + (size_t)(kt * 16 + ln) * D_SZ;
        if (kt + 1 < S_SZ / 16)
            __builtin_prefetch(Kb + (size_t)((kt + 1) * 16 + ln) * D_SZ, 0, 3);
        // B fragment of K^T chunk == row-major K rows with the same per-lane packing
        v16h bk0 = load_frag_f32(krow, kb0);
        v16h bk1 = load_frag_f32(krow, 32 + kb0);
        v8f c = {};
        c = __builtin_amdgcn_wmma_f32_16x16x32_f16(false, aq0, false, bk0, (short)0, c, false, false);
        c = __builtin_amdgcn_wmma_f32_16x16x32_f16(false, aq1, false, bk1, (short)0, c, false, false);
        // C layout: lanes 0-15 N=lane (VGPR i -> M=i), lanes 16-31 N=lane-16 (M=i+8)
#pragma unroll
        for (int i = 0; i < 8; ++i) {
            int m = i + (lo ? 0 : 8);
            probs[m * ROWSTRIDE + kt * 16 + ln] = (_Float16)c[i];
        }
    }

    // ---- Phase 2: softmax over each row of 2048 (2 lanes per row) ----
    {
        const int half = lane >> 4;
        _Float16* rowp = probs + ln * ROWSTRIDE + half * 1024;
        float m = -1e30f;
        for (int j = 0; j < 1024; j += 8) {
            v8h x = *(const v8h*)(rowp + j);
#pragma unroll
            for (int e = 0; e < 8; ++e) m = fmaxf(m, (float)x[e]);
        }
        m = fmaxf(m, __shfl_xor(m, 16, 32));
        float lsum = 0.f;
        for (int j = 0; j < 1024; j += 8) {
            v8h x = *(const v8h*)(rowp + j);
#pragma unroll
            for (int e = 0; e < 8; ++e) lsum += __expf((float)x[e] - m);
        }
        lsum += __shfl_xor(lsum, 16, 32);
        const float inv = 1.0f / lsum;
        for (int j = 0; j < 1024; j += 8) {
            v8h x = *(const v8h*)(rowp + j);
#pragma unroll
            for (int e = 0; e < 8; ++e) x[e] = (_Float16)(__expf((float)x[e] - m) * inv);
            *(v8h*)(rowp + j) = x;
        }
    }

    // ---- Phase 2b: write attention output (fp32, coalesced float4) ----
    for (int r = 0; r < 16; ++r) {
        float* arow = attn + ((size_t)(b * S_SZ) + qb * 16 + r) * S_SZ;
        const _Float16* prow = probs + r * ROWSTRIDE;
        for (int j = lane * 4; j < S_SZ; j += 32 * 4) {
            float4 o;
            o.x = (float)prow[j + 0]; o.y = (float)prow[j + 1];
            o.z = (float)prow[j + 2]; o.w = (float)prow[j + 3];
            *(float4*)(arow + j) = o;
        }
    }

    // ---- Phase 3: context = P @ V  (M=16, N=64 in 4 tiles, K=2048 in 64 chunks) ----
    v8f acc[4] = {v8f{}, v8f{}, v8f{}, v8f{}};
    for (int kc = 0; kc < S_SZ / 32; ++kc) {
        const _Float16* pr = probs + ln * ROWSTRIDE + kc * 32;
        v8h plo = *(const v8h*)(pr + kb0);
        v8h phi = *(const v8h*)(pr + kb0 + 16);
        v16h ap;
#pragma unroll
        for (int e = 0; e < 8; ++e) { ap[e] = plo[e]; ap[8 + e] = phi[e]; }
#pragma unroll
        for (int nt = 0; nt < 4; ++nt) {
            // B fragment of V chunk: lane holds column d = nt*16+ln, K-subset {kb0..kb0+7, kb0+16..}
            const float* vcol = Vb + (size_t)(kc * 32 + kb0) * D_SZ + nt * 16 + ln;
            v16h bv;
#pragma unroll
            for (int e = 0; e < 8; ++e) {
                bv[e]     = (_Float16)vcol[e * D_SZ];
                bv[8 + e] = (_Float16)vcol[(16 + e) * D_SZ];
            }
            acc[nt] = __builtin_amdgcn_wmma_f32_16x16x32_f16(false, ap, false, bv, (short)0,
                                                             acc[nt], false, false);
        }
    }

    // ---- Store context + residual Q ----
#pragma unroll
    for (int nt = 0; nt < 4; ++nt) {
#pragma unroll
        for (int i = 0; i < 8; ++i) {
            int m = i + (lo ? 0 : 8);
            size_t idx = ((size_t)(b * S_SZ) + qb * 16 + m) * D_SZ + nt * 16 + ln;
            ctx[idx] = acc[nt][i] + q[idx];
        }
    }
}

extern "C" void kernel_launch(void* const* d_in, const int* in_sizes, int n_in,
                              void* d_out, int out_size, void* d_ws, size_t ws_size,
                              hipStream_t stream) {
    const float* q = (const float*)d_in[0];
    const float* k = (const float*)d_in[1];
    const float* v = (const float*)d_in[2];
    float* out = (float*)d_out;

    // 263168 B of dynamic LDS (> 64 KB default cap) — raise the limit.
    (void)hipFuncSetAttribute(reinterpret_cast<const void*>(attn_fused_kernel),
                              hipFuncAttributeMaxDynamicSharedMemorySize, LDS_BYTES);

    const int qblocks = B_SZ * (S_SZ / 16);           // 4096 waves of work
    dim3 grid(qblocks / WAVES_PER_WG);                // 1024 workgroups
    dim3 block(32 * WAVES_PER_WG);                    // 128 threads = 4 waves (wave32)
    attn_fused_kernel<<<grid, block, LDS_BYTES, stream>>>(q, k, v, out);
}
